// TextAttention_29738353557822
// MI455X (gfx1250) — matrix operand order repaired and checked by reading, not compiled, not run on hardware
//
#include <hip/hip_runtime.h>

// Problem constants from the reference (B=1).
#define T_LEN   4096
#define D_DIM   2048
#define H_HEADS 16
#define G_GROUPS 4
#define K_DIM   128
#define SCALE_F 0.08838834764831845f  // 1/sqrt(128)

typedef __attribute__((ext_vector_type(16))) _Float16 v16h;
typedef __attribute__((ext_vector_type(8)))  _Float16 v8h;
typedef __attribute__((ext_vector_type(4)))  _Float16 v4h;
typedef __attribute__((ext_vector_type(8)))  float    v8f;

__device__ __forceinline__ v8f wmma16x16x32(v16h a, v16h b, v8f c) {
  // (neg_a, A, neg_b, B, c_mod, C, reuse_a, reuse_b)
  return __builtin_amdgcn_wmma_f32_16x16x32_f16(false, a, false, b, (short)0, c,
                                                false, false);
}

// A-matrix fragment (16x32 f16, MxK) from a row-major LDS tile.
// ISA 7.12.2: lane&15 = row M; lanes 0-15 hold K {0..7,16..23}, lanes 16-31
// hold K {8..15,24..31}.  Two aligned 16B ds loads per lane.
__device__ __forceinline__ v16h frag_a_lds(const _Float16* base, int lda, int lane) {
  const int m = lane & 15, half = lane >> 4;
  const _Float16* row = base + m * lda + half * 8;
  union { v16h v; v8h h[2]; } u;
  u.h[0] = *(const v8h*)(row);
  u.h[1] = *(const v8h*)(row + 16);
  return u.v;
}

// B-matrix fragment (32x16 f16, KxN) from an LDS tile stored TRANSPOSED as
// Bt[n][k] (row stride ldb).  ISA: lane&15 = column N; lanes 0-15 hold
// K {0..7,16..23}, lanes 16-31 hold K {8..15,24..31} (matches frag_a_lds).
__device__ __forceinline__ v16h frag_b_lds(const _Float16* base, int ldb, int lane) {
  const int n = lane & 15, khalf = lane >> 4;
  const _Float16* p = base + n * ldb + khalf * 8;
  union { v16h v; v8h h[2]; } u;
  u.h[0] = *(const v8h*)(p);
  u.h[1] = *(const v8h*)(p + 16);
  return u.v;
}

// ---------------------------------------------------------------------------
// Elementwise fp32 -> f16 (vectorized x4).  n must be a multiple of 4.
// ---------------------------------------------------------------------------
__global__ __launch_bounds__(256)
void cvt_f16(const float* __restrict__ x, _Float16* __restrict__ y, int n4) {
  const int i = blockIdx.x * 256 + threadIdx.x;
  if (i < n4) {
    const float4 v = ((const float4*)x)[i];
    v4h h;
    h[0] = (_Float16)v.x; h[1] = (_Float16)v.y;
    h[2] = (_Float16)v.z; h[3] = (_Float16)v.w;
    ((v4h*)y)[i] = h;
  }
}

// ---------------------------------------------------------------------------
// W (Kd x N, f32, row-major) -> Wt (N x Kd, f16).  32x32 LDS tile transpose,
// block (32,8), grid (Kd/32, N/32).  Both global sides coalesced.
// ---------------------------------------------------------------------------
__global__ __launch_bounds__(256)
void transpose_f16(const float* __restrict__ W, _Float16* __restrict__ Wt,
                   int Kd, int N) {
  __shared__ float tile[32][33];
  const int k0 = blockIdx.x * 32, n0 = blockIdx.y * 32;
  const int tx = threadIdx.x, ty = threadIdx.y;
#pragma unroll
  for (int i = 0; i < 32; i += 8)
    tile[ty + i][tx] = W[(size_t)(k0 + ty + i) * N + (n0 + tx)];
  __syncthreads();
#pragma unroll
  for (int i = 0; i < 32; i += 8)
    Wt[(size_t)(n0 + ty + i) * Kd + (k0 + tx)] = (_Float16)tile[tx][ty + i];
}

// ---------------------------------------------------------------------------
// C = A(MxKd,f16) * Bt(NxKd,f16)^T, f32 out.  128xBN tile per workgroup,
// 4 waves x 32 rows, BK=32.  LDS ping-pong + register-staged software
// pipeline: global_load_b128 for tile kt+1 issues right after the barrier,
// the matching ds_store happens at the top of iteration kt+1 — so the
// s_wait_loadcnt lands a full WMMA phase after the loads were issued and
// the WMMA chain never stalls on HBM.  M%128==N%BN==Kd%32==0.
// ---------------------------------------------------------------------------
template <int BN>
__global__ __launch_bounds__(128)
void gemm_f16wmma(const _Float16* __restrict__ A, const _Float16* __restrict__ Bt,
                  float* __restrict__ C, int M, int N, int Kd) {
  constexpr int NT  = BN / 16;               // n-subtiles per wave
  constexpr int BCH = (BN * 32 / 8) / 128;   // B staging chunks per thread
  __shared__ __align__(16) _Float16 lA[2][128][32];
  __shared__ __align__(16) _Float16 lB[2][BN][32];
  const int tid  = threadIdx.x;
  const int lane = tid & 31;
  const int wave = tid >> 5;
  const int tileM = blockIdx.x * 128;
  const int tileN = blockIdx.y * BN;

  v8f acc[2][NT];
#pragma unroll
  for (int i = 0; i < 2; ++i)
#pragma unroll
    for (int j = 0; j < NT; ++j)
#pragma unroll
      for (int r = 0; r < 8; ++r) acc[i][j][r] = 0.0f;

  v8h rA[4], rB[BCH];  // register staging (16B chunks)

  auto gload = [&](int k0) {
#pragma unroll
    for (int it = 0; it < 4; ++it) {
      const int idx = tid + it * 128;           // 0..511 over 128x32 A tile
      const int m = idx >> 2, c = (idx & 3) * 8;
      rA[it] = *(const v8h*)&A[(size_t)(tileM + m) * Kd + k0 + c];
    }
#pragma unroll
    for (int it = 0; it < BCH; ++it) {
      const int idx = tid + it * 128;           // over BNx32 B tile
      const int n = idx >> 2, c = (idx & 3) * 8;
      rB[it] = *(const v8h*)&Bt[(size_t)(tileN + n) * Kd + k0 + c];
    }
  };
  auto lstore = [&](int buf) {
#pragma unroll
    for (int it = 0; it < 4; ++it) {
      const int idx = tid + it * 128;
      const int m = idx >> 2, c = (idx & 3) * 8;
      *(v8h*)&lA[buf][m][c] = rA[it];
    }
#pragma unroll
    for (int it = 0; it < BCH; ++it) {
      const int idx = tid + it * 128;
      const int n = idx >> 2, c = (idx & 3) * 8;
      *(v8h*)&lB[buf][n][c] = rB[it];
    }
  };

  const int nK = Kd / 32;
  gload(0);
  for (int kt = 0; kt < nK; ++kt) {
    const int buf = kt & 1;
    lstore(buf);            // stores regs loaded during previous compute phase
    __syncthreads();
    if (kt + 1 < nK) gload((kt + 1) * 32);   // issue next tile's HBM loads now
    const v16h af0 = frag_a_lds(&lA[buf][wave * 32][0], 32, lane);
    const v16h af1 = frag_a_lds(&lA[buf][wave * 32 + 16][0], 32, lane);
#pragma unroll
    for (int j = 0; j < NT; ++j) {
      const v16h bf = frag_b_lds(&lB[buf][j * 16][0], 32, lane);
      acc[0][j] = wmma16x16x32(af0, bf, acc[0][j]);
      acc[1][j] = wmma16x16x32(af1, bf, acc[1][j]);
    }
  }

  // C/D layout: row = (lane>>4)*8 + vgpr, col = lane&15
  const int mb = tileM + wave * 32 + (lane >> 4) * 8;
  const int nb = tileN + (lane & 15);
#pragma unroll
  for (int i = 0; i < 2; ++i)
#pragma unroll
    for (int j = 0; j < NT; ++j)
#pragma unroll
      for (int r = 0; r < 8; ++r)
        C[(size_t)(mb + i * 16 + r) * N + nb + j * 16] = acc[i][j][r];
}

// ---------------------------------------------------------------------------
// In-place RMS-norm (+ scale) followed by RoPE on one (t, head) row of K=128.
// block = 128 threads; grid = (T, numHeads).  All math f32.
// ---------------------------------------------------------------------------
__global__ __launch_bounds__(128)
void rmsnorm_rope(float* __restrict__ x, const float* __restrict__ sin_tk,
                  const float* __restrict__ cos_tk, const float* __restrict__ scale,
                  int Hh) {
  const int t = blockIdx.x;
  const int h = blockIdx.y;
  const int k = threadIdx.x;
  float* row = x + ((size_t)t * Hh + h) * K_DIM;
  const float v = row[k];

  __shared__ float red[128];
  __shared__ float nrm[128];
  red[k] = v * v;
  __syncthreads();
#pragma unroll
  for (int s = 64; s > 0; s >>= 1) {
    if (k < s) red[k] += red[k + s];
    __syncthreads();
  }
  const float inv = rsqrtf(red[0] * (1.0f / 128.0f) + 1e-6f);
  nrm[k] = v * inv * scale[k];
  __syncthreads();

  const int p  = k & 63;
  const float sn = sin_tk[(size_t)t * 64 + p];
  const float cs = cos_tk[(size_t)t * 64 + p];
  const float x1 = nrm[p];
  const float x2 = nrm[p + 64];
  row[k] = (k < 64) ? (x1 * cs - x2 * sn) : (x2 * cs + x1 * sn);
}

// ---------------------------------------------------------------------------
// Causal GQA flash attention.  grid = (T/64, H); block = 128 (4 waves).
// Each wave owns 16 query rows of head h; workgroup shares K/V LDS tiles of
// 32 keys from group g = h/4.  Output written as f16 (feeds the Wo GEMM).
// All waves run the same number of key blocks (fully-masked blocks give
// exp()=0) so barriers stay uniform.
// ---------------------------------------------------------------------------
__global__ __launch_bounds__(128)
void attention_fwd(const float* __restrict__ q, const float* __restrict__ kmat,
                   const float* __restrict__ vmat, _Float16* __restrict__ attn) {
  __shared__ __align__(16) _Float16 lK[32][128];    // [key][feat]  (= Bt for QK^T)
  __shared__ __align__(16) _Float16 lVt[128][32];   // [feat][key]  (= Bt for P@V)
  __shared__ __align__(16) _Float16 lP[4][16][32];  // per-wave P transpose scratch

  const int tid  = threadIdx.x;
  const int lane = tid & 31;
  const int wave = tid >> 5;
  const int h = blockIdx.y;
  const int g = h >> 2;                       // R = H/G = 4
  const int qb = blockIdx.x * 64;
  const int qrow0 = qb + wave * 16;
  const int half = lane >> 4;
  const int mrow_off = half * 8;
  const int ncol = lane & 15;

  // Q fragments (A layout) straight from global, fold in 1/sqrt(K).
  v16h qf[4];
  {
    const int m = lane & 15;
    const float* qrow = q + ((size_t)(qrow0 + m) * H_HEADS + h) * K_DIM;
#pragma unroll
    for (int c = 0; c < 4; ++c) {
      v16h a;
#pragma unroll
      for (int e = 0; e < 16; ++e) {
        const int kk = c * 32 + (e >> 3) * 16 + half * 8 + (e & 7);
        a[e] = (_Float16)(qrow[kk] * SCALE_F);
      }
      qf[c] = a;
    }
  }

  v8f o[8];
  float mstat[8], lstat[8];
#pragma unroll
  for (int ft = 0; ft < 8; ++ft)
#pragma unroll
    for (int r = 0; r < 8; ++r) o[ft][r] = 0.0f;
#pragma unroll
  for (int r = 0; r < 8; ++r) { mstat[r] = -1e30f; lstat[r] = 0.0f; }

  const int nkb = (qb >> 5) + 2;  // key blocks of 32 covering keys <= qb+63
  for (int kb = 0; kb < nkb; ++kb) {
    const int s0 = kb * 32;
    // Stage K and V (vectorized float4 -> packed f16): 32 keys x 32 chunks.
    for (int idx = tid; idx < 32 * 32; idx += 128) {
      const int s = idx >> 5, f = (idx & 31) * 4;
      const size_t src = ((size_t)(s0 + s) * G_GROUPS + g) * K_DIM + f;
      const float4 kv = *(const float4*)&kmat[src];
      v4h kh;
      kh[0] = (_Float16)kv.x; kh[1] = (_Float16)kv.y;
      kh[2] = (_Float16)kv.z; kh[3] = (_Float16)kv.w;
      *(v4h*)&lK[s][f] = kh;
      const float4 vv = *(const float4*)&vmat[src];
      lVt[f + 0][s] = (_Float16)vv.x;
      lVt[f + 1][s] = (_Float16)vv.y;
      lVt[f + 2][s] = (_Float16)vv.z;
      lVt[f + 3][s] = (_Float16)vv.w;
    }
    __syncthreads();

    // S = Q * K^T for 16 queries x 32 keys (2 n-tiles x 4 k-chunks)
    v8f sc[2];
#pragma unroll
    for (int j = 0; j < 2; ++j)
#pragma unroll
      for (int r = 0; r < 8; ++r) sc[j][r] = 0.0f;
#pragma unroll
    for (int c = 0; c < 4; ++c)
#pragma unroll
      for (int j = 0; j < 2; ++j) {
        const v16h bf = frag_b_lds(&lK[j * 16][c * 32], 128, lane);
        sc[j] = wmma16x16x32(qf[c], bf, sc[j]);
      }

    // Causal mask (finite -inf surrogate keeps exp() exactly 0, no NaNs).
#pragma unroll
    for (int j = 0; j < 2; ++j)
#pragma unroll
      for (int r = 0; r < 8; ++r) {
        const int tq = qrow0 + mrow_off + r;
        const int sk = s0 + j * 16 + ncol;
        if (sk > tq) sc[j][r] = -1e30f;
      }

    // Online softmax; row m = half*8 + r lives across the 16 lanes of this
    // half-wave, so xor masks 1/2/4/8 reduce exactly one row.
#pragma unroll
    for (int r = 0; r < 8; ++r) {
      float vmax = fmaxf(sc[0][r], sc[1][r]);
      vmax = fmaxf(vmax, __shfl_xor(vmax, 1, 32));
      vmax = fmaxf(vmax, __shfl_xor(vmax, 2, 32));
      vmax = fmaxf(vmax, __shfl_xor(vmax, 4, 32));
      vmax = fmaxf(vmax, __shfl_xor(vmax, 8, 32));
      const float newM = fmaxf(mstat[r], vmax);
      const float e0 = __expf(sc[0][r] - newM);
      const float e1 = __expf(sc[1][r] - newM);
      sc[0][r] = e0; sc[1][r] = e1;
      float rowsum = e0 + e1;
      rowsum += __shfl_xor(rowsum, 1, 32);
      rowsum += __shfl_xor(rowsum, 2, 32);
      rowsum += __shfl_xor(rowsum, 4, 32);
      rowsum += __shfl_xor(rowsum, 8, 32);
      const float alpha = __expf(mstat[r] - newM);
      lstat[r] = lstat[r] * alpha + rowsum;
      mstat[r] = newM;
#pragma unroll
      for (int ft = 0; ft < 8; ++ft) o[ft][r] *= alpha;
    }

    // Transpose P from C/D layout to A layout via per-wave LDS scratch.
#pragma unroll
    for (int j = 0; j < 2; ++j)
#pragma unroll
      for (int r = 0; r < 8; ++r)
        lP[wave][mrow_off + r][j * 16 + ncol] = (_Float16)sc[j][r];
    __syncthreads();

    const v16h pf = frag_a_lds(&lP[wave][0][0], 32, lane);
#pragma unroll
    for (int ft = 0; ft < 8; ++ft) {
      const v16h bf = frag_b_lds(&lVt[ft * 16][0], 32, lane);
      o[ft] = wmma16x16x32(pf, bf, o[ft]);
    }
    __syncthreads();  // protect lK/lVt/lP before next block's staging
  }

#pragma unroll
  for (int r = 0; r < 8; ++r) {
    const float inv = 1.0f / lstat[r];
    const int tq = qrow0 + mrow_off + r;
#pragma unroll
    for (int ft = 0; ft < 8; ++ft)
      attn[(size_t)tq * (H_HEADS * K_DIM) + h * K_DIM + ft * 16 + ncol] =
          (_Float16)(o[ft][r] * inv);
  }
}

// ---------------------------------------------------------------------------
extern "C" void kernel_launch(void* const* d_in, const int* in_sizes, int n_in,
                              void* d_out, int out_size, void* d_ws, size_t ws_size,
                              hipStream_t stream) {
  const float* hidden  = (const float*)d_in[0];
  const float* sin_tk  = (const float*)d_in[1];
  const float* cos_tk  = (const float*)d_in[2];
  /* d_in[3] = mask: causal, computed analytically */
  const float* Wq      = (const float*)d_in[4];
  const float* Wk      = (const float*)d_in[5];
  const float* Wv      = (const float*)d_in[6];
  const float* Wo      = (const float*)d_in[7];
  const float* q_scale = (const float*)d_in[8];
  const float* k_scale = (const float*)d_in[9];
  float* out = (float*)d_out;

  // Workspace carve (all 16B aligned): ~100 MB total.
  char* p = (char*)d_ws;
  _Float16* Xh   = (_Float16*)p; p += (size_t)T_LEN * D_DIM * 2;            // 16 MB
  _Float16* Wqt  = (_Float16*)p; p += (size_t)2048 * 2048 * 2;              //  8 MB
  _Float16* Wkt  = (_Float16*)p; p += (size_t)512 * 2048 * 2;               //  2 MB
  _Float16* Wvt  = (_Float16*)p; p += (size_t)512 * 2048 * 2;               //  2 MB
  _Float16* Wot  = (_Float16*)p; p += (size_t)2048 * 2048 * 2;              //  8 MB
  float*    q_raw = (float*)p;   p += (size_t)T_LEN * 2048 * 4;             // 32 MB
  float*    k_raw = (float*)p;   p += (size_t)T_LEN * 512 * 4;              //  8 MB
  float*    v_raw = (float*)p;   p += (size_t)T_LEN * 512 * 4;              //  8 MB
  _Float16* attn  = (_Float16*)p;                                            // 16 MB

  const dim3 blk(128);
  const dim3 tblk(32, 8);

  // Pre-pass: f16 copies of activations / pre-transposed f16 weights.
  cvt_f16<<<dim3((T_LEN * D_DIM / 4 + 255) / 256), dim3(256), 0, stream>>>(
      hidden, Xh, T_LEN * D_DIM / 4);
  transpose_f16<<<dim3(D_DIM / 32, 2048 / 32), tblk, 0, stream>>>(Wq, Wqt, D_DIM, 2048);
  transpose_f16<<<dim3(D_DIM / 32, 512 / 32),  tblk, 0, stream>>>(Wk, Wkt, D_DIM, 512);
  transpose_f16<<<dim3(D_DIM / 32, 512 / 32),  tblk, 0, stream>>>(Wv, Wvt, D_DIM, 512);
  transpose_f16<<<dim3(2048 / 32, D_DIM / 32), tblk, 0, stream>>>(Wo, Wot, 2048, D_DIM);

  // Projections (f16 WMMA, f32 accumulate/out).
  gemm_f16wmma<128><<<dim3(T_LEN / 128, 2048 / 128), blk, 0, stream>>>(
      Xh, Wqt, q_raw, T_LEN, 2048, D_DIM);
  gemm_f16wmma<64><<<dim3(T_LEN / 128, 512 / 64), blk, 0, stream>>>(
      Xh, Wkt, k_raw, T_LEN, 512, D_DIM);
  gemm_f16wmma<64><<<dim3(T_LEN / 128, 512 / 64), blk, 0, stream>>>(
      Xh, Wvt, v_raw, T_LEN, 512, D_DIM);

  rmsnorm_rope<<<dim3(T_LEN, H_HEADS), blk, 0, stream>>>(q_raw, sin_tk, cos_tk,
                                                         q_scale, H_HEADS);
  rmsnorm_rope<<<dim3(T_LEN, G_GROUPS), blk, 0, stream>>>(k_raw, sin_tk, cos_tk,
                                                          k_scale, G_GROUPS);

  attention_fwd<<<dim3(T_LEN / 64, H_HEADS), blk, 0, stream>>>(q_raw, k_raw,
                                                               v_raw, attn);

  // Output projection: attn(f16) @ Wo -> out(f32).
  gemm_f16wmma<128><<<dim3(T_LEN / 128, D_DIM / 128), blk, 0, stream>>>(
      attn, Wot, out, T_LEN, D_DIM, 2048);
}
